// GatedAttention_27247272525828
// MI455X (gfx1250) — matrix operand order repaired and checked by reading, not compiled, not run on hardware
//
#include <hip/hip_runtime.h>

typedef __attribute__((ext_vector_type(16))) __bf16 v16bf;
typedef __attribute__((ext_vector_type(8)))  __bf16 v8bf;
typedef __attribute__((ext_vector_type(8)))  float  v8f;

namespace {
constexpr int kT0 = 512;
constexpr int kT1 = 1536;
constexpr int kT  = 2048;
constexpr int kW  = 2048;   // WIDTH
constexpr int kN  = 16;     // query heads
constexpr int kK  = 4;      // kv heads
constexpr int kH  = 128;    // head dim
constexpr float kNeg = -2.3819763e38f;
constexpr float kQScale = 0.08838834764831845f; // 1/sqrt(128)
}

// ---------------------------------------------------------------------------
// WMMA fragment helpers (wave32, bf16 16x16x32, per CDNA5 ISA 7.12.2)
// ---------------------------------------------------------------------------

// A matrix 16x32 (MxK), row-major source with leading dim ld (elements).
// lane<16: M=lane, K in {0..7} U {16..23}; lane>=16: M=lane-16, K in {8..15} U {24..31}
__device__ __forceinline__ v16bf load_frag_a(const __bf16* p, int ld) {
  int lane = threadIdx.x & 31;
  int row  = lane & 15;
  int kb   = (lane & 16) ? 8 : 0;
  const __bf16* q = p + (size_t)row * ld + kb;
  v8bf lo = *(const v8bf*)q;
  v8bf hi = *(const v8bf*)(q + 16);
  v16bf a;
#pragma unroll
  for (int i = 0; i < 8; i++) { a[i] = lo[i]; a[8 + i] = hi[i]; }
  return a;
}

// B matrix 32x16 (KxN), loaded from transposed source Bt[n][k] with leading dim ld.
// lane<16: N=lane, K in {0..7} U {16..23}; lane>=16: N=lane-16, K in {8..15} U {24..31}
__device__ __forceinline__ v16bf load_frag_b(const __bf16* p, int ld) {
  int lane = threadIdx.x & 31;
  int n    = lane & 15;
  int kb   = (lane & 16) ? 8 : 0;
  const __bf16* q = p + (size_t)n * ld + kb;
  v8bf lo = *(const v8bf*)q;
  v8bf hi = *(const v8bf*)(q + 16);
  v16bf b;
#pragma unroll
  for (int i = 0; i < 8; i++) { b[i] = lo[i]; b[8 + i] = hi[i]; }
  return b;
}

__device__ __forceinline__ v8f wmma_bf16(v16bf a, v16bf b, v8f c) {
  return __builtin_amdgcn_wmma_f32_16x16x32_bf16(
      /*neg_a=*/false, a, /*neg_b=*/false, b,
      /*c_mod=*/(short)0, c, /*reuse_a=*/false, /*reuse_b=*/false);
}

// ---------------------------------------------------------------------------
// gfx1250 async copy: one b128 global->LDS per lane, tracked by ASYNCcnt.
// Low 32 bits of a generic shared pointer are the LDS byte offset (ISA 10.2).
// ---------------------------------------------------------------------------
__device__ __forceinline__ void async_copy_b128(__bf16* lds_dst, const __bf16* gsrc) {
  unsigned loff = (unsigned)(uintptr_t)lds_dst;
  unsigned long long ga = (unsigned long long)(uintptr_t)gsrc;
  asm volatile("global_load_async_to_lds_b128 %0, %1, off"
               :: "v"(loff), "v"(ga) : "memory");
}
__device__ __forceinline__ void wait_async_le(int n) {
  if (n == 0) asm volatile("s_wait_asynccnt 0" ::: "memory");
  else        asm volatile("s_wait_asynccnt 1" ::: "memory");
}

// ---------------------------------------------------------------------------
// 1. token-mask + f32 -> bf16 activation prep
// ---------------------------------------------------------------------------
__global__ void prep_x_kernel(const float* __restrict__ x0, const float* __restrict__ x1,
                              const unsigned char* __restrict__ mask, __bf16* __restrict__ xb) {
  int t = blockIdx.x;  // 0..kT-1
  __shared__ int anyf;
  if (threadIdx.x == 0) anyf = 0;
  __syncthreads();
  const unsigned int* mrow = (const unsigned int*)(mask + (size_t)t * kT);
  unsigned int acc = 0;
  for (int i = threadIdx.x; i < kT / 4; i += blockDim.x) acc |= mrow[i];
  if (acc) atomicOr(&anyf, 1);
  __syncthreads();
  float f = anyf ? 1.0f : 0.0f;
  const float* src = (t < kT0) ? (x0 + (size_t)t * kW) : (x1 + (size_t)(t - kT0) * kW);
  for (int i = threadIdx.x; i < kW; i += blockDim.x)
    xb[(size_t)t * kW + i] = (__bf16)(src[i] * f);
}

// ---------------------------------------------------------------------------
// 2. weight conversion + transpose: src (M,R,C) f32 -> dst (M,C,R) bf16
// ---------------------------------------------------------------------------
__global__ void transpose_bf16_kernel(const float* __restrict__ src, __bf16* __restrict__ dst,
                                      int M, int R, int C) {
  size_t total = (size_t)M * R * C;
  for (size_t i = (size_t)blockIdx.x * blockDim.x + threadIdx.x; i < total;
       i += (size_t)gridDim.x * blockDim.x) {
    size_t m = i / ((size_t)R * C);
    size_t rem = i - m * (size_t)R * C;
    size_t r = rem / C;
    size_t c = rem - r * C;
    dst[m * (size_t)R * C + c * (size_t)R + r] = (__bf16)src[i];
  }
}

// ---------------------------------------------------------------------------
// 3. projection GEMM (64 tokens x 256 cols x K=2048), async-LDS staged A tile,
//    RMSNorm/RoPE epilogue.  mode 0 = QG (q + sigmoid gate), mode 1 = KV.
// ---------------------------------------------------------------------------
__global__ void proj_kernel(const __bf16* __restrict__ xb,
                            const __bf16* __restrict__ WTa, const __bf16* __restrict__ WTb,
                            long strideA, long strideB,
                            const float* __restrict__ nscale,
                            const int* __restrict__ positions,
                            int tOff, int mode,
                            __bf16* __restrict__ out_a,   // QG: q_bf (N,T,H)   KV: k_bf (K,T,H)
                            __bf16* __restrict__ out_b,   // QG: gate (T,N,H)   KV: vT (K,H,T)
                            float* __restrict__ ck, float* __restrict__ cv) {
  __shared__ __align__(16) __bf16 atile[2][64 * 32];  // double-buffered A tile
  __shared__ __align__(16) float  ctile[64 * 128];    // q/k half for norm+rope
  __shared__ float rstd[64];

  int head  = blockIdx.y;
  int tglob = tOff + blockIdx.x * 64;
  int w     = threadIdx.x >> 5;
  int lane  = threadIdx.x & 31;
  int cb    = w * 32;

  const __bf16* Wp0;
  const __bf16* Wp1;
  {
    int c0 = cb, c1 = cb + 16;
    Wp0 = (c0 < 128) ? (WTa + (size_t)head * strideA + (size_t)c0 * kW)
                     : (WTb + (size_t)head * strideB + (size_t)(c0 - 128) * kW);
    Wp1 = (c1 < 128) ? (WTa + (size_t)head * strideA + (size_t)c1 * kW)
                     : (WTb + (size_t)head * strideB + (size_t)(c1 - 128) * kW);
  }

  // async staging: each thread owns one 16B chunk of the 64x32 bf16 tile
  int srow   = threadIdx.x >> 2;
  int schunk = (threadIdx.x & 3) * 8;
  const __bf16* Sp = xb + (size_t)(tglob + srow) * kW + schunk;

  async_copy_b128(&atile[0][srow * 32 + schunk], Sp);  // prologue: k=0

  v8f acc[4][2] = {{{}, {}}, {{}, {}}, {{}, {}}, {{}, {}}};

  constexpr int nk = kW / 32;
  for (int s = 0; s < nk; s++) {
    int buf = s & 1;
    bool pre = (s + 1 < nk);
    if (pre) async_copy_b128(&atile[buf ^ 1][srow * 32 + schunk], Sp + (s + 1) * 32);
    wait_async_le(pre ? 1 : 0);
    __syncthreads();  // whole A tile visible to all waves

    int k = s * 32;
    __builtin_prefetch(Wp0 + k + 512, 0, 0);
    v16bf b0 = load_frag_b(Wp0 + k, kW);
    v16bf b1 = load_frag_b(Wp1 + k, kW);
#pragma unroll
    for (int rt = 0; rt < 4; rt++) {
      v16bf a = load_frag_a(&atile[buf][rt * 16 * 32], 32);
      acc[rt][0] = wmma_bf16(a, b0, acc[rt][0]);
      acc[rt][1] = wmma_bf16(a, b1, acc[rt][1]);
    }
    __syncthreads();  // all reads of buf done before it is refilled
  }

  int r0  = (lane & 16) ? 8 : 0;
  int col = lane & 15;

  if (cb < 128) {
    // q/k half -> LDS for cross-lane rmsnorm + rope
#pragma unroll
    for (int rt = 0; rt < 4; rt++)
#pragma unroll
      for (int r = 0; r < 8; r++) {
        ctile[(rt * 16 + r0 + r) * 128 + cb + col]      = acc[rt][0][r];
        ctile[(rt * 16 + r0 + r) * 128 + cb + 16 + col] = acc[rt][1][r];
      }
  } else {
    // gate / v half: write directly from accumulators (C-fragment layout)
#pragma unroll
    for (int rt = 0; rt < 4; rt++)
#pragma unroll
      for (int ct = 0; ct < 2; ct++)
#pragma unroll
        for (int r = 0; r < 8; r++) {
          int t = tglob + rt * 16 + r0 + r;
          int h = (cb - 128) + ct * 16 + col;
          float raw = acc[rt][ct][r];
          if (mode == 0) {
            float sig = 1.0f / (1.0f + __expf(-raw));
            out_b[((size_t)t * kN + head) * kH + h] = (__bf16)sig;
          } else {
            out_b[((size_t)head * kH + h) * kT + t] = (__bf16)raw;  // vT (K,H,T)
            cv[((size_t)t * kK + head) * kH + h] = raw;
          }
        }
  }
  __syncthreads();

  // per-row rms over 128 cols
  if (threadIdx.x < 64) {
    int row = threadIdx.x;
    float s = 0.f;
    for (int c = 0; c < 128; c++) { float v = ctile[row * 128 + c]; s += v * v; }
    rstd[row] = rsqrtf(s * (1.0f / 128.0f) + 1e-6f);
  }
  __syncthreads();

  // normalize in place (each element depends only on itself)
  for (int e = threadIdx.x; e < 64 * 128; e += blockDim.x) {
    int row = e >> 7, c = e & 127;
    ctile[e] = ctile[e] * rstd[row] * (1.0f + nscale[c]);
  }
  __syncthreads();

  // rope + scale + outputs
  float qscale = (mode == 0) ? kQScale : 1.0f;
  for (int e = threadIdx.x; e < 64 * 128; e += blockDim.x) {
    int row = e >> 7, c = e & 127;
    int t = tglob + row;
    float val;
    if (c < 64) {  // rope over first 64 (half = 32)
      int j = c & 31;
      float ang = (float)positions[t] * __powf(1.0e6f, -(float)j * (1.0f / 32.0f));
      float cs = __cosf(ang), sn = __sinf(ang);
      float x1v = ctile[row * 128 + j];
      float x2v = ctile[row * 128 + 32 + j];
      val = (c < 32) ? (x1v * cs - x2v * sn) : (x2v * cs + x1v * sn);
    } else {
      val = ctile[row * 128 + c];
    }
    val *= qscale;
    out_a[((size_t)head * kT + t) * kH + c] = (__bf16)val;
    if (mode == 1) ck[((size_t)t * kK + head) * kH + c] = val;
  }
}

// ---------------------------------------------------------------------------
// 4. flash attention: one (head, 64-query tile) per block, 8 waves
// ---------------------------------------------------------------------------
__global__ void attn_kernel(const __bf16* __restrict__ qbf,   // (N,T,H)
                            const __bf16* __restrict__ kbf,   // (K,T,H)
                            const __bf16* __restrict__ vT,    // (K,H,T)
                            const __bf16* __restrict__ gate,  // (T,N,H) sigmoid applied
                            __bf16* __restrict__ encb) {      // (T, N*H)
  __shared__ __align__(16) float  slog[64 * 32];
  __shared__ __align__(16) __bf16 probs[64 * 32];
  __shared__ float rowm[64], rowl[64], rowf[64];

  int n     = blockIdx.y;
  int kh    = n >> 2;  // N/K = 4
  int qbase = blockIdx.x * 64;
  int w     = threadIdx.x >> 5;
  int lane  = threadIdx.x & 31;
  int rt    = w & 3;          // row tile (logits + enc)
  int cw    = w >> 2;         // logits key sub-tile (0/1)
  int ecb   = (w >> 2) * 64;  // enc column base (0 or 64)

  // preload this wave's Q fragments (16 rows x 128 h = 4 frags)
  v16bf aq[4];
  const __bf16* Qp = qbf + ((size_t)n * kT + qbase + rt * 16) * kH;
#pragma unroll
  for (int kk = 0; kk < 4; kk++) aq[kk] = load_frag_a(Qp + kk * 32, kH);

  v8f acc[4] = {{}, {}, {}, {}};
  if (threadIdx.x < 64) { rowm[threadIdx.x] = -__builtin_inff(); rowl[threadIdx.x] = 0.f; }
  __syncthreads();

  int nS = (qbase + 64) / 32;  // causal: keys 0 .. qbase+63
  for (int sb = 0; sb < nS; sb++) {
    // ---- logits 16x16 tile per wave (K = H = 128) ----
    v8f c = {};
    const __bf16* Kp = kbf + ((size_t)kh * kT + sb * 32 + cw * 16) * kH;
#pragma unroll
    for (int kk = 0; kk < 4; kk++) {
      v16bf b = load_frag_b(Kp + kk * 32, kH);
      c = wmma_bf16(aq[kk], b, c);
    }
    {
      int r0  = (lane & 16) ? 8 : 0;
      int col = lane & 15;
#pragma unroll
      for (int r = 0; r < 8; r++) {
        int qrow = rt * 16 + r0 + r;
        int key  = sb * 32 + cw * 16 + col;
        float v  = (key <= qbase + qrow) ? c[r] : kNeg;
        slog[qrow * 32 + cw * 16 + col] = v;
      }
    }
    __syncthreads();

    // ---- online softmax row stats + bf16 probs ----
    if (threadIdx.x < 64) {
      int row = threadIdx.x;
      float m_old = rowm[row];
      float tm = -__builtin_inff();
      for (int j = 0; j < 32; j++) tm = fmaxf(tm, slog[row * 32 + j]);
      float m_new = fmaxf(m_old, tm);
      float f = __expf(m_old - m_new);
      float s = 0.f;
      for (int j = 0; j < 32; j++) {
        float p = __expf(slog[row * 32 + j] - m_new);
        probs[row * 32 + j] = (__bf16)p;
        s += p;
      }
      rowl[row] = rowl[row] * f + s;
      rowm[row] = m_new;
      rowf[row] = f;
    }
    __syncthreads();

    // ---- rescale accumulators, then enc += P(16x32) @ V(32x16) ----
    {
      int r0 = (lane & 16) ? 8 : 0;
      v16bf ap = load_frag_a(&probs[rt * 16 * 32], 32);
#pragma unroll
      for (int j = 0; j < 4; j++) {
#pragma unroll
        for (int r = 0; r < 8; r++) acc[j][r] *= rowf[rt * 16 + r0 + r];
        const __bf16* Vp = vT + ((size_t)kh * kH + ecb + j * 16) * kT + sb * 32;
        v16bf b = load_frag_b(Vp, kT);
        acc[j] = wmma_bf16(ap, b, acc[j]);
      }
    }
    __syncthreads();
  }

  // ---- epilogue: 1/l, sigmoid gate, write bf16 enc ----
  {
    int r0  = (lane & 16) ? 8 : 0;
    int col = lane & 15;
#pragma unroll
    for (int j = 0; j < 4; j++) {
      int hb = ecb + j * 16 + col;
#pragma unroll
      for (int r = 0; r < 8; r++) {
        int row = rt * 16 + r0 + r;
        int t = qbase + row;
        float inv = 1.0f / rowl[row];
        float g = (float)gate[((size_t)t * kN + n) * kH + hb];
        encb[(size_t)t * (kN * kH) + n * kH + hb] = (__bf16)(acc[j][r] * inv * g);
      }
    }
  }
}

// ---------------------------------------------------------------------------
// 5. output projection: enc(64 x 2048) @ Wo(2048 x 128-col slab) per block,
//    async-LDS staged A tile.
// ---------------------------------------------------------------------------
__global__ void out_kernel(const __bf16* __restrict__ encb,  // (T, N*H)
                           const __bf16* __restrict__ WoT,   // (WIDTH, N*H) transposed
                           float* __restrict__ out, int tOff) {
  __shared__ __align__(16) __bf16 atile[2][64 * 32];

  int w    = threadIdx.x >> 5;
  int lane = threadIdx.x & 31;
  int rowbase = blockIdx.x * 64;              // stream-local row base
  int colbase = blockIdx.y * 128 + w * 16;    // this wave's output column tile

  const __bf16* Wp = WoT + (size_t)colbase * (kN * kH);

  int srow   = threadIdx.x >> 2;
  int schunk = (threadIdx.x & 3) * 8;
  const __bf16* Sp = encb + (size_t)(tOff + rowbase + srow) * (kN * kH) + schunk;

  async_copy_b128(&atile[0][srow * 32 + schunk], Sp);

  v8f acc[4] = {{}, {}, {}, {}};

  constexpr int nk = (kN * kH) / 32;
  for (int s = 0; s < nk; s++) {
    int buf = s & 1;
    bool pre = (s + 1 < nk);
    if (pre) async_copy_b128(&atile[buf ^ 1][srow * 32 + schunk], Sp + (s + 1) * 32);
    wait_async_le(pre ? 1 : 0);
    __syncthreads();

    int k = s * 32;
    __builtin_prefetch(Wp + k + 512, 0, 0);
    v16bf b = load_frag_b(Wp + k, kN * kH);
#pragma unroll
    for (int rt = 0; rt < 4; rt++) {
      v16bf a = load_frag_a(&atile[buf][rt * 16 * 32], 32);
      acc[rt] = wmma_bf16(a, b, acc[rt]);
    }
    __syncthreads();
  }

  int r0  = (lane & 16) ? 8 : 0;
  int col = lane & 15;
#pragma unroll
  for (int rt = 0; rt < 4; rt++)
#pragma unroll
    for (int r = 0; r < 8; r++)
      out[(size_t)(rowbase + rt * 16 + r0 + r) * kW + colbase + col] = acc[rt][r];
}

// ---------------------------------------------------------------------------
// host launch
// ---------------------------------------------------------------------------
extern "C" void kernel_launch(void* const* d_in, const int* in_sizes, int n_in,
                              void* d_out, int out_size, void* d_ws, size_t ws_size,
                              hipStream_t stream) {
  (void)in_sizes; (void)n_in; (void)out_size; (void)ws_size;
  const float* x0   = (const float*)d_in[0];
  const float* x1   = (const float*)d_in[1];
  const int*   pos  = (const int*)d_in[2];
  const unsigned char* mask = (const unsigned char*)d_in[3];
  const float* qg_w0 = (const float*)d_in[4];
  const float* k_w0  = (const float*)d_in[5];
  const float* v_w0  = (const float*)d_in[6];
  const float* qn0   = (const float*)d_in[7];
  const float* kn0   = (const float*)d_in[8];
  const float* o_w0  = (const float*)d_in[9];
  const float* qg_w1 = (const float*)d_in[10];
  const float* k_w1  = (const float*)d_in[11];
  const float* v_w1  = (const float*)d_in[12];
  const float* qn1   = (const float*)d_in[13];
  const float* kn1   = (const float*)d_in[14];
  const float* o_w1  = (const float*)d_in[15];

  char* ws = (char*)d_ws;
  size_t off = 0;
  auto take = [&](size_t bytes) { char* p = ws + off; off += (bytes + 255) & ~(size_t)255; return p; };

  __bf16* xb    = (__bf16*)take((size_t)kT * kW * 2);                // 8.4 MB
  __bf16* WqgT0 = (__bf16*)take((size_t)kN * 256 * kW * 2);          // 16.8 MB
  __bf16* WqgT1 = (__bf16*)take((size_t)kN * 256 * kW * 2);
  __bf16* WkT0  = (__bf16*)take((size_t)kK * kH * kW * 2);           // 2.1 MB
  __bf16* WkT1  = (__bf16*)take((size_t)kK * kH * kW * 2);
  __bf16* WvT0  = (__bf16*)take((size_t)kK * kH * kW * 2);
  __bf16* WvT1  = (__bf16*)take((size_t)kK * kH * kW * 2);
  __bf16* WoT0  = (__bf16*)take((size_t)kW * kW * 2);                // 8.4 MB
  __bf16* WoT1  = (__bf16*)take((size_t)kW * kW * 2);
  __bf16* qbf   = (__bf16*)take((size_t)kN * kT * kH * 2);           // 8.4 MB
  __bf16* kbf   = (__bf16*)take((size_t)kK * kT * kH * 2);           // 2.1 MB
  __bf16* vTb   = (__bf16*)take((size_t)kK * kH * kT * 2);
  __bf16* gateb = (__bf16*)take((size_t)kT * kN * kH * 2);           // 8.4 MB
  __bf16* encb  = (__bf16*)take((size_t)kT * kN * kH * 2);

  float* out = (float*)d_out;
  float* out0 = out;                                   // (T0, W)
  float* out1 = out + (size_t)kT0 * kW;                // (T1, W)
  float* ck   = out + (size_t)kT * kW;                 // (T, K, H)
  float* cv   = ck + (size_t)kT * kK * kH;             // (T, K, H)

  // weight conversion + transpose to bf16
  transpose_bf16_kernel<<<2048, 256, 0, stream>>>(qg_w0, WqgT0, kN, kW, 256);
  transpose_bf16_kernel<<<2048, 256, 0, stream>>>(qg_w1, WqgT1, kN, kW, 256);
  transpose_bf16_kernel<<<1024, 256, 0, stream>>>(k_w0, WkT0, kK, kW, kH);
  transpose_bf16_kernel<<<1024, 256, 0, stream>>>(k_w1, WkT1, kK, kW, kH);
  transpose_bf16_kernel<<<1024, 256, 0, stream>>>(v_w0, WvT0, kK, kW, kH);
  transpose_bf16_kernel<<<1024, 256, 0, stream>>>(v_w1, WvT1, kK, kW, kH);
  transpose_bf16_kernel<<<2048, 256, 0, stream>>>(o_w0, WoT0, 1, kN * kH, kW);
  transpose_bf16_kernel<<<2048, 256, 0, stream>>>(o_w1, WoT1, 1, kN * kH, kW);

  // token mask + activation bf16
  prep_x_kernel<<<kT, 256, 0, stream>>>(x0, x1, mask, xb);

  long sQG = (long)256 * kW;
  long sKV = (long)kH * kW;
  // stream 0: QG + KV projections (M = 64 token tiles)
  proj_kernel<<<dim3(kT0 / 64, kN), 256, 0, stream>>>(
      xb, WqgT0, WqgT0 + (size_t)128 * kW, sQG, sQG, qn0, pos, 0, 0,
      qbf, gateb, nullptr, nullptr);
  proj_kernel<<<dim3(kT0 / 64, kK), 256, 0, stream>>>(
      xb, WkT0, WvT0, sKV, sKV, kn0, pos, 0, 1,
      kbf, vTb, ck, cv);
  // stream 1
  proj_kernel<<<dim3(kT1 / 64, kN), 256, 0, stream>>>(
      xb, WqgT1, WqgT1 + (size_t)128 * kW, sQG, sQG, qn1, pos, kT0, 0,
      qbf, gateb, nullptr, nullptr);
  proj_kernel<<<dim3(kT1 / 64, kK), 256, 0, stream>>>(
      xb, WkT1, WvT1, sKV, sKV, kn1, pos, kT0, 1,
      kbf, vTb, ck, cv);

  // attention (causal, GQA 16q/4kv, sigmoid gate fused into enc)
  attn_kernel<<<dim3(kT / 64, kN), 256, 0, stream>>>(qbf, kbf, vTb, gateb, encb);

  // output projections per stream
  out_kernel<<<dim3(kT0 / 64, kW / 128), 256, 0, stream>>>(encb, WoT0, out0, 0);
  out_kernel<<<dim3(kT1 / 64, kW / 128), 256, 0, stream>>>(encb, WoT1, out1, kT0);
}
